// DecoderAttModule_71141838291260
// MI455X (gfx1250) — matrix operand order repaired and checked by reading, not compiled, not run on hardware
//
#include <hip/hip_runtime.h>
#include <hip/hip_bf16.h>

// Problem constants (from reference)
#define B_  256
#define N_  36
#define F_  2048
#define L_  26
#define A_  512
#define E_  1024
#define D_  1024
#define V_  10000
#define T_  25
#define ARW 0.005f

typedef __attribute__((ext_vector_type(16))) __bf16 bf16x16;
typedef __attribute__((ext_vector_type(8)))  float  f32x8;

// ---------------------------------------------------------------------------
// WMMA GEMM:  C[M x N] = A[M x K](fp32,row-major) @ W[N x K](bf16,row-major)^T
//             + bias[N]; optional per-row mask zeroing (for masked preds).
//
// Block tile 128(M) x 128(N), K step 32, double-buffered LDS.
// 8 waves (256 thr) in a 4(M) x 2(N) grid; each wave owns a 32x64 sub-tile
// = 2x4 v_wmma_f32_16x16x32_bf16 fragments (8 WMMA / K-chunk / wave).
// Weight tile is streamed with global_load_async_to_lds_b128 (ASYNCcnt);
// activation tile is converted fp32->bf16 through VGPRs into LDS.
// Out-of-range rows are CLAMPED (not branched) so EXEC stays all-ones for
// WMMA; garbage only reaches accumulator rows/cols the epilogue never stores.
// ---------------------------------------------------------------------------
#define BM 128
#define BN 128
#define BK 32

__global__ __launch_bounds__(256) void k_gemm(
    const float* __restrict__ A, int lda,
    const __hip_bfloat16* __restrict__ W, int ldb,
    const float* __restrict__ bias,
    float* __restrict__ C, long ldc,
    int M, int N, int K,
    const int* __restrict__ row_mask)
{
  __shared__ __align__(16) __hip_bfloat16 As[2][BM * BK];
  __shared__ __align__(16) __hip_bfloat16 Bs[2][BN * BK];

  const int tid  = threadIdx.x;
  const int wave = tid >> 5;
  const int lane = tid & 31;
  const int wr   = wave >> 1;      // 0..3 : wave row  (M, 32 each)
  const int wcn  = wave & 1;       // 0..1 : wave col  (N, 64 each)
  const int m_block = blockIdx.y * BM;
  const int n_block = blockIdx.x * BN;

  const int l16 = lane & 15;
  const int lh  = lane >> 4;

  f32x8 acc[2][4];
#pragma unroll
  for (int a = 0; a < 2; ++a)
#pragma unroll
    for (int b = 0; b < 4; ++b)
#pragma unroll
      for (int i = 0; i < 8; ++i) acc[a][b][i] = 0.0f;

  // ---- staging assignments: each thread owns a 16-element row segment ----
  const int arow = tid >> 1;            // 0..127
  const int acg  = (tid & 1) * 16;      // 0,16
  const int brow = arow;
  const int bcg  = acg;

  const int a_gr  = m_block + arow;
  const int a_grc = a_gr < M ? a_gr : (M - 1);          // clamp, no divergence
  const float* a_src_row = A + (long)a_grc * lda;

  const int b_gn  = n_block + brow;
  const int b_gnc = b_gn < N ? b_gn : (N - 1);          // clamp, no divergence
  const int b_row_byte = (int)((long)b_gnc * ldb * 2) + bcg * 2;

  auto fillA = [&](int buf, int k0) {
    const float* src = a_src_row + k0 + acg;
    const float4 f0 = *(const float4*)(src + 0);
    const float4 f1 = *(const float4*)(src + 4);
    const float4 f2 = *(const float4*)(src + 8);
    const float4 f3 = *(const float4*)(src + 12);
    union { __hip_bfloat16 h[16]; uint4 u[2]; } pk;
    pk.h[0]  = __float2bfloat16(f0.x); pk.h[1]  = __float2bfloat16(f0.y);
    pk.h[2]  = __float2bfloat16(f0.z); pk.h[3]  = __float2bfloat16(f0.w);
    pk.h[4]  = __float2bfloat16(f1.x); pk.h[5]  = __float2bfloat16(f1.y);
    pk.h[6]  = __float2bfloat16(f1.z); pk.h[7]  = __float2bfloat16(f1.w);
    pk.h[8]  = __float2bfloat16(f2.x); pk.h[9]  = __float2bfloat16(f2.y);
    pk.h[10] = __float2bfloat16(f2.z); pk.h[11] = __float2bfloat16(f2.w);
    pk.h[12] = __float2bfloat16(f3.x); pk.h[13] = __float2bfloat16(f3.y);
    pk.h[14] = __float2bfloat16(f3.z); pk.h[15] = __float2bfloat16(f3.w);
    uint4* dst = (uint4*)(&As[buf][arow * BK + acg]);
    dst[0] = pk.u[0]; dst[1] = pk.u[1];
  };

  auto fillB = [&](int buf, int k0) {
    // 32 bytes of one weight row -> LDS, DMA'd without touching VGPR data.
    // offset:16 applies to BOTH the LDS and global side (ISA 10. async ops).
    const unsigned lds_off =
        (unsigned)(unsigned long long)(&Bs[buf][brow * BK + bcg]);
    const int gbyte = b_row_byte + k0 * 2;
    asm volatile(
        "global_load_async_to_lds_b128 %0, %1, %2\n\t"
        "global_load_async_to_lds_b128 %0, %1, %2 offset:16"
        :
        : "v"(lds_off), "v"(gbyte), "s"(W)
        : "memory");
  };

  // ---- prologue: fill buffer 0 ----
  fillA(0, 0);
  fillB(0, 0);
  asm volatile("s_wait_asynccnt 0x0" ::: "memory");
  __syncthreads();

  const int nchunks = K >> 5;
  for (int kc = 0; kc < nchunks; ++kc) {
    const int cur  = kc & 1;
    const bool more = (kc + 1) < nchunks;
    if (more) {             // prefetch next K-chunk into the other buffer
      fillA(cur ^ 1, (kc + 1) << 5);
      fillB(cur ^ 1, (kc + 1) << 5);
    }

    // ---- per-lane fragments (ISA 7.12.2 layouts) ----
    union { uint4 u[2]; bf16x16 v; } afr[2], bfr[4];
#pragma unroll
    for (int fm = 0; fm < 2; ++fm) {
      const int row = wr * 32 + fm * 16 + l16;
      const int kb  = lh * 8;  // lanes 0-15: K 0..7 & 16..23 ; 16-31: 8..15 & 24..31
      afr[fm].u[0] = *(const uint4*)(&As[cur][row * BK + kb]);
      afr[fm].u[1] = *(const uint4*)(&As[cur][row * BK + kb + 16]);
    }
#pragma unroll
    for (int fn = 0; fn < 4; ++fn) {
      const int row = wcn * 64 + fn * 16 + l16;  // W row == B column
      const int kb  = lh * 8;                    // lanes 0-15: K 0..7 & 16..23 ; 16-31: 8..15 & 24..31
      bfr[fn].u[0] = *(const uint4*)(&Bs[cur][row * BK + kb]);
      bfr[fn].u[1] = *(const uint4*)(&Bs[cur][row * BK + kb + 16]);
    }
#pragma unroll
    for (int fm = 0; fm < 2; ++fm)
#pragma unroll
      for (int fn = 0; fn < 4; ++fn)
        acc[fm][fn] = __builtin_amdgcn_wmma_f32_16x16x32_bf16(
            false, afr[fm].v, false, bfr[fn].v, (short)0, acc[fm][fn],
            false, false);

    if (more) asm volatile("s_wait_asynccnt 0x0" ::: "memory");
    __syncthreads();
  }

  // ---- epilogue: bias add, optional row mask, bounds-guarded store ----
#pragma unroll
  for (int fm = 0; fm < 2; ++fm) {
#pragma unroll
    for (int fn = 0; fn < 4; ++fn) {
      const int col = n_block + wcn * 64 + fn * 16 + l16;
      if (col < N) {
        const float bv = bias[col];
#pragma unroll
        for (int i = 0; i < 8; ++i) {
          const int row = m_block + wr * 32 + fm * 16 + lh * 8 + i;
          if (row < M) {
            float v = acc[fm][fn][i] + bv;
            if (row_mask && !row_mask[row]) v = 0.0f;
            C[(long)row * ldc + col] = v;
          }
        }
      }
    }
  }
}

// ---------------------------------------------------------------------------
// Setup / elementwise kernels
// ---------------------------------------------------------------------------
__global__ void k_order(const int* __restrict__ sizes, int* __restrict__ order,
                        int* __restrict__ dec_len)
{
  __shared__ int ss[B_];
  const int b = threadIdx.x;
  ss[b] = sizes[b];   // sizes is (B,1)
  __syncthreads();
  const int s = ss[b];
  int rank = 0;
  for (int j = 0; j < B_; ++j) {
    const int sj = ss[j];
    if (sj > s || (sj == s && j < b)) rank++;  // stable descending
  }
  order[rank]   = b;
  dec_len[rank] = s - 1;
}

__global__ void k_gather_fs(const float* __restrict__ feats,
                            const int* __restrict__ order,
                            float* __restrict__ fs)
{
  const int idx = blockIdx.x * blockDim.x + threadIdx.x;
  if (idx >= B_ * N_ * F_) return;
  const int r    = idx / (N_ * F_);
  const int rest = idx - r * (N_ * F_);
  fs[idx] = feats[(long)order[r] * (N_ * F_) + rest];
}

__global__ void k_favg(const float* __restrict__ fs, float* __restrict__ favg)
{
  const int idx = blockIdx.x * blockDim.x + threadIdx.x;
  if (idx >= B_ * F_) return;
  const int b = idx / F_, f = idx - b * F_;
  float s = 0.f;
  const float* p = fs + (long)b * N_ * F_ + f;
  for (int n = 0; n < N_; ++n) s += p[n * F_];
  favg[idx] = s * (1.0f / N_);
}

__global__ void k_conv_bf16(const float* __restrict__ src, int scols,
                            __hip_bfloat16* __restrict__ dst, int dcols,
                            int coff, int total)
{
  const int idx = blockIdx.x * blockDim.x + threadIdx.x;
  if (idx >= total) return;
  const int r = idx / scols, c = idx - r * scols;
  dst[(long)r * dcols + coff + c] = __float2bfloat16(src[idx]);
}

__global__ void k_copycols(const float* __restrict__ src, int scols,
                           float* __restrict__ dst, int dcols, int coff,
                           int total)
{
  const int idx = blockIdx.x * blockDim.x + threadIdx.x;
  if (idx >= total) return;
  const int r = idx / scols, c = idx - r * scols;
  dst[(long)r * dcols + coff + c] = src[idx];
}

__global__ void k_embed(const float* __restrict__ emb,
                        const int* __restrict__ seqs,
                        const int* __restrict__ order,
                        float* __restrict__ x1, int t)
{
  const int idx = blockIdx.x * blockDim.x + threadIdx.x;
  if (idx >= B_ * E_) return;
  const int b = idx >> 10, e = idx & (E_ - 1);
  const int tok = seqs[order[b] * L_ + t];
  x1[(long)b * 5120 + 3072 + e] = emb[(long)tok * E_ + e];
}

__global__ void k_masks(const int* __restrict__ dec_len, int* __restrict__ mask,
                        float* __restrict__ bszf)
{
  const int t = blockIdx.x, b = threadIdx.x;
  const int act = (t < dec_len[b]) ? 1 : 0;
  mask[t * B_ + b] = act;
  __shared__ int red[B_];
  red[b] = act; __syncthreads();
  for (int s = B_ / 2; s > 0; s >>= 1) {
    if (b < s) red[b] += red[b + s];
    __syncthreads();
  }
  if (b == 0) bszf[t] = (float)red[0];
}

__device__ __forceinline__ float sigf(float x) { return 1.0f / (1.0f + __expf(-x)); }

__global__ void k_lstm_gate(const float* __restrict__ g,
                            const float* __restrict__ c_in,
                            float* __restrict__ h_out,
                            float* __restrict__ c_out)
{
  const int idx = blockIdx.x * blockDim.x + threadIdx.x;
  if (idx >= B_ * D_) return;
  const int b = idx / D_, d = idx - b * D_;
  const float* gb = g + (long)b * 4 * D_;
  const float gi = gb[d], gf = gb[D_ + d], gg = gb[2 * D_ + d], go = gb[3 * D_ + d];
  const float c2 = sigf(gf) * c_in[idx] + sigf(gi) * tanhf(gg);
  c_out[idx] = c2;
  h_out[idx] = sigf(go) * tanhf(c2);
}

__global__ void k_attention(const float* __restrict__ q,
                            const float* __restrict__ img_att,
                            const float* __restrict__ att_w,
                            const float* __restrict__ att_b,
                            const float* __restrict__ fs,
                            float* __restrict__ aw)
{
  const int b = blockIdx.x, tid = threadIdx.x;   // 256 threads
  __shared__ float sq[A_], sw[A_], red[256], salpha[N_];
  for (int i = tid; i < A_; i += 256) { sq[i] = q[b * A_ + i]; sw[i] = att_w[i]; }
  __syncthreads();
  const float bt = att_b[0];
  for (int n = 0; n < N_; ++n) {
    const float* ia = img_att + ((long)b * N_ + n) * A_;
    float p = 0.f;
    for (int a = tid; a < A_; a += 256) {
      float v = sq[a] + ia[a];
      v = v > 0.f ? v : 0.f;
      p += v * sw[a];
    }
    red[tid] = p; __syncthreads();
    for (int s = 128; s > 0; s >>= 1) {
      if (tid < s) red[tid] += red[tid + s];
      __syncthreads();
    }
    if (tid == 0) salpha[n] = red[0] + bt;
    __syncthreads();
  }
  if (tid == 0) {
    float m = salpha[0];
    for (int n = 1; n < N_; ++n) m = fmaxf(m, salpha[n]);
    float s = 0.f;
    for (int n = 0; n < N_; ++n) { salpha[n] = __expf(salpha[n] - m); s += salpha[n]; }
    const float inv = 1.0f / s;
    for (int n = 0; n < N_; ++n) salpha[n] *= inv;
  }
  __syncthreads();
  for (int f = tid; f < F_; f += 256) {
    float acc = 0.f;
    const float* fb = fs + (long)b * N_ * F_ + f;
    for (int n = 0; n < N_; ++n) acc += salpha[n] * fb[n * F_];
    aw[(long)b * F_ + f] = acc;
  }
}

__global__ void k_ar_loss(const float* __restrict__ arl_out,
                          const float* __restrict__ prev_h1,
                          const int* __restrict__ mask_t,
                          const float* __restrict__ bszf_t,
                          float* __restrict__ loss, int t)
{
  const int b = blockIdx.x, tid = threadIdx.x;
  float p = 0.f;
  const float* x = arl_out + (long)b * D_;
  const float* y = prev_h1 + (long)b * D_;
  for (int d = tid; d < D_; d += 256) {
    const float df = x[d] - y[d];
    p += df * df;
  }
  __shared__ float red[256];
  red[tid] = p; __syncthreads();
  for (int s = 128; s > 0; s >>= 1) {
    if (tid < s) red[tid] += red[tid + s];
    __syncthreads();
  }
  if (tid == 0 && t > 0 && mask_t[b]) {
    float bsz = bszf_t[0];
    if (bsz < 1.f) bsz = 1.f;
    atomicAdd(loss, red[0] * ARW / bsz);
  }
}

__global__ void k_commit(float* h1, float* c1, float* h2, float* c2,
                         float* arh, float* arc, float* prev_h1,
                         const float* h1n, const float* c1n,
                         const float* h2n, const float* c2n,
                         const float* arhn, const float* arcn,
                         const int* __restrict__ mask_t)
{
  const int idx = blockIdx.x * blockDim.x + threadIdx.x;
  if (idx >= B_ * D_) return;
  const int b = idx / D_;
  if (mask_t[b]) {
    h1[idx] = h1n[idx];   c1[idx] = c1n[idx];
    h2[idx] = h2n[idx];   c2[idx] = c2n[idx];
    arh[idx] = arhn[idx]; arc[idx] = arcn[idx];
    prev_h1[idx] = h1n[idx];
  }
}

// ---------------------------------------------------------------------------
// Host launcher
// ---------------------------------------------------------------------------
static inline int cdiv(long a, long b) { return (int)((a + b - 1) / b); }

extern "C" void kernel_launch(void* const* d_in, const int* in_sizes, int n_in,
                              void* d_out, int out_size, void* d_ws, size_t ws_size,
                              hipStream_t stream)
{
  // ---- inputs (setup_inputs order) ----
  const float* feats   = (const float*)d_in[0];
  const int*   seqs    = (const int*)  d_in[1];
  const int*   sizes   = (const int*)  d_in[2];
  const float* emb     = (const float*)d_in[3];
  const float* td_wih  = (const float*)d_in[4];
  const float* td_whh  = (const float*)d_in[5];
  const float* td_b    = (const float*)d_in[6];
  const float* l_wih   = (const float*)d_in[7];
  const float* l_whh   = (const float*)d_in[8];
  const float* l_b     = (const float*)d_in[9];
  const float* attf_w  = (const float*)d_in[10];
  const float* attf_b  = (const float*)d_in[11];
  const float* attd_w  = (const float*)d_in[12];
  const float* attd_b  = (const float*)d_in[13];
  const float* att_w   = (const float*)d_in[14];
  const float* att_b   = (const float*)d_in[15];
  const float* out_w   = (const float*)d_in[16];
  const float* out_b   = (const float*)d_in[17];
  const float* ar_wih  = (const float*)d_in[18];
  const float* ar_whh  = (const float*)d_in[19];
  const float* ar_b    = (const float*)d_in[20];
  const float* arl_w   = (const float*)d_in[21];
  const float* arl_b   = (const float*)d_in[22];

  float* preds = (float*)d_out;                       // (B, T, V)
  float* loss  = (float*)d_out + (size_t)B_ * T_ * V_;

  // ---- workspace carve-up ----
  size_t off = 0;
  auto alloc = [&](size_t bytes) -> char* {
    off = (off + 255) & ~(size_t)255;
    char* p = (char*)d_ws + off;
    off += bytes;
    return p;
  };
  int*   order   = (int*)  alloc(B_ * 4);
  int*   dec_len = (int*)  alloc(B_ * 4);
  int*   mask    = (int*)  alloc((size_t)T_ * B_ * 4);
  float* bszf    = (float*)alloc(T_ * 4);
  float* fs      = (float*)alloc((size_t)B_ * N_ * F_ * 4);
  float* favg    = (float*)alloc((size_t)B_ * F_ * 4);
  float* img_att = (float*)alloc((size_t)B_ * N_ * A_ * 4);
  __hip_bfloat16* W1   = (__hip_bfloat16*)alloc((size_t)4096 * 5120 * 2);  // [td_wih|td_whh]
  __hip_bfloat16* W2   = (__hip_bfloat16*)alloc((size_t)4096 * 4096 * 2);  // [lang_wih|lang_whh]
  __hip_bfloat16* War  = (__hip_bfloat16*)alloc((size_t)4096 * 2048 * 2);  // [ar_wih|ar_whh]
  __hip_bfloat16* Wout = (__hip_bfloat16*)alloc((size_t)V_ * D_ * 2);
  __hip_bfloat16* Wattf= (__hip_bfloat16*)alloc((size_t)A_ * F_ * 2);
  __hip_bfloat16* Wattd= (__hip_bfloat16*)alloc((size_t)A_ * D_ * 2);
  __hip_bfloat16* Warl = (__hip_bfloat16*)alloc((size_t)D_ * D_ * 2);
  float* states = (float*)alloc((size_t)7 * B_ * D_ * 4);  // contiguous for one memset
  float* h1 = states + 0 * B_ * D_;
  float* c1 = states + 1 * (size_t)B_ * D_;
  float* h2 = states + 2 * (size_t)B_ * D_;
  float* c2 = states + 3 * (size_t)B_ * D_;
  float* arh = states + 4 * (size_t)B_ * D_;
  float* arc = states + 5 * (size_t)B_ * D_;
  float* prev_h1 = states + 6 * (size_t)B_ * D_;
  float* h1n  = (float*)alloc((size_t)B_ * D_ * 4);
  float* c1n  = (float*)alloc((size_t)B_ * D_ * 4);
  float* h2n  = (float*)alloc((size_t)B_ * D_ * 4);
  float* c2n  = (float*)alloc((size_t)B_ * D_ * 4);
  float* arhn = (float*)alloc((size_t)B_ * D_ * 4);
  float* arcn = (float*)alloc((size_t)B_ * D_ * 4);
  float* x1   = (float*)alloc((size_t)B_ * 5120 * 4);
  float* x2   = (float*)alloc((size_t)B_ * 4096 * 4);
  float* xar  = (float*)alloc((size_t)B_ * 2048 * 4);
  float* g1   = (float*)alloc((size_t)B_ * 4096 * 4);
  float* g2   = (float*)alloc((size_t)B_ * 4096 * 4);
  float* gar  = (float*)alloc((size_t)B_ * 4096 * 4);
  float* qbuf = (float*)alloc((size_t)B_ * A_ * 4);
  float* aw   = (float*)alloc((size_t)B_ * F_ * 4);
  float* arl_out = (float*)alloc((size_t)B_ * D_ * 4);
  (void)ws_size; (void)in_sizes; (void)n_in; (void)out_size;

  const int TB = 256;
  auto gemm = [&](const float* Ap, int lda, const __hip_bfloat16* Wp, int ldb,
                  const float* bp, float* Cp, long ldc, int M, int N, int K,
                  const int* rm) {
    dim3 grid(cdiv(N, BN), cdiv(M, BM));
    k_gemm<<<grid, 256, 0, stream>>>(Ap, lda, Wp, ldb, bp, Cp, ldc, M, N, K, rm);
  };
  auto conv = [&](const float* s, int sc, __hip_bfloat16* d, int dc, int co, int rows) {
    int total = rows * sc;
    k_conv_bf16<<<cdiv(total, TB), TB, 0, stream>>>(s, sc, d, dc, co, total);
  };
  auto copyc = [&](const float* s, int sc, float* d, int dc, int co, int rows) {
    int total = rows * sc;
    k_copycols<<<cdiv(total, TB), TB, 0, stream>>>(s, sc, d, dc, co, total);
  };

  // ---- one-time setup ----
  k_order<<<1, B_, 0, stream>>>(sizes, order, dec_len);
  k_gather_fs<<<cdiv((long)B_ * N_ * F_, TB), TB, 0, stream>>>(feats, order, fs);
  k_favg<<<cdiv((long)B_ * F_, TB), TB, 0, stream>>>(fs, favg);
  k_masks<<<T_, B_, 0, stream>>>(dec_len, mask, bszf);

  conv(td_wih, 4096, W1, 5120, 0, 4096);
  conv(td_whh, 1024, W1, 5120, 4096, 4096);
  conv(l_wih, 3072, W2, 4096, 0, 4096);
  conv(l_whh, 1024, W2, 4096, 3072, 4096);
  conv(ar_wih, 1024, War, 2048, 0, 4096);
  conv(ar_whh, 1024, War, 2048, 1024, 4096);
  conv(out_w, 1024, Wout, 1024, 0, V_);
  conv(attf_w, 2048, Wattf, 2048, 0, A_);
  conv(attd_w, 1024, Wattd, 1024, 0, A_);
  conv(arl_w, 1024, Warl, 1024, 0, 1024);

  hipMemsetAsync(states, 0, (size_t)7 * B_ * D_ * 4, stream);
  hipMemsetAsync(loss, 0, 4, stream);

  // img_att[B*N, A] = fs[B*N, F] @ attf_w^T + attf_b
  gemm(fs, F_, Wattf, F_, attf_b, img_att, A_, B_ * N_, A_, F_, nullptr);

  // ---- time loop ----
  for (int t = 0; t < T_; ++t) {
    const int* mt = mask + (size_t)t * B_;

    // x1 = [h2 | favg | e_t | h1]  (5120)
    copyc(h2, D_, x1, 5120, 0, B_);
    copyc(favg, F_, x1, 5120, 1024, B_);
    k_embed<<<cdiv((long)B_ * E_, TB), TB, 0, stream>>>(emb, seqs, order, x1, t);
    copyc(h1, D_, x1, 5120, 4096, B_);

    gemm(x1, 5120, W1, 5120, td_b, g1, 4096, B_, 4096, 5120, nullptr);
    k_lstm_gate<<<cdiv((long)B_ * D_, TB), TB, 0, stream>>>(g1, c1, h1n, c1n);

    // attention
    gemm(h1n, D_, Wattd, D_, attd_b, qbuf, A_, B_, A_, D_, nullptr);
    k_attention<<<B_, 256, 0, stream>>>(qbuf, img_att, att_w, att_b, fs, aw);

    // x2 = [aw | h1n | h2]  (4096)
    copyc(aw, F_, x2, 4096, 0, B_);
    copyc(h1n, D_, x2, 4096, 2048, B_);
    copyc(h2, D_, x2, 4096, 3072, B_);
    gemm(x2, 4096, W2, 4096, l_b, g2, 4096, B_, 4096, 4096, nullptr);
    k_lstm_gate<<<cdiv((long)B_ * D_, TB), TB, 0, stream>>>(g2, c2, h2n, c2n);

    // AR-Net LSTM: xar = [h1n | arh]  (2048)
    copyc(h1n, D_, xar, 2048, 0, B_);
    copyc(arh, D_, xar, 2048, 1024, B_);
    gemm(xar, 2048, War, 2048, ar_b, gar, 4096, B_, 4096, 2048, nullptr);
    k_lstm_gate<<<cdiv((long)B_ * D_, TB), TB, 0, stream>>>(gar, arc, arhn, arcn);

    // arl projection + loss
    gemm(arhn, D_, Warl, D_, arl_b, arl_out, D_, B_, D_, D_, nullptr);
    k_ar_loss<<<B_, 256, 0, stream>>>(arl_out, prev_h1, mt, bszf + t, loss, t);

    // masked output projection, written straight to d_out[b][t][:]
    gemm(h2n, D_, Wout, D_, out_b, preds + (size_t)t * V_, (long)T_ * V_,
         B_, V_, D_, mt);

    // masked state commit
    k_commit<<<cdiv((long)B_ * D_, TB), TB, 0, stream>>>(
        h1, c1, h2, c2, arh, arc, prev_h1, h1n, c1n, h2n, c2n, arhn, arcn, mt);
  }
}
